// AbstractionLayer_87222195847181
// MI455X (gfx1250) — matrix operand order repaired and checked by reading, not compiled, not run on hardware
//
#include <hip/hip_runtime.h>
#include <hip/hip_bf16.h>

// ---------------------------------------------------------------------------
// Problem constants (from reference): B=8, N=16384, M=512, R=32
// MLP: 3->64->64->128->1024, affine+relu each layer, max over R, out (B,M,1024,1)
// ---------------------------------------------------------------------------
#define NPTS   16384
#define NB     8
#define NM     512
#define NR     32
#define NGRP   (NB * NM)          // 4096 groups

typedef __attribute__((ext_vector_type(16))) _Float16 v16h;
typedef __attribute__((ext_vector_type(8)))  _Float16 v8h;
typedef __attribute__((ext_vector_type(8)))  float    v8f;
typedef __attribute__((ext_vector_type(4)))  int      v4i;

#if defined(__has_builtin)
#  if __has_builtin(__builtin_amdgcn_global_load_async_to_lds_b128)
#    define HAVE_ASYNC_LDS_BUILTIN 1
#  endif
#  if __has_builtin(__builtin_amdgcn_s_wait_asynccnt)
#    define HAVE_WAIT_ASYNC_BUILTIN 1
#  endif
#endif

// ---------------------------------------------------------------------------
// Kernel 1: convert weights to f16 (w0 padded K:3->32)
// ---------------------------------------------------------------------------
__global__ void prep_weights_kernel(const float* __restrict__ w0,
                                    const float* __restrict__ w1,
                                    const float* __restrict__ w2,
                                    const float* __restrict__ w3,
                                    _Float16* __restrict__ w0p,   // [64][32]
                                    _Float16* __restrict__ w1p,   // [64][64]
                                    _Float16* __restrict__ w2p,   // [128][64]
                                    _Float16* __restrict__ w3p) { // [1024][128]
  int i = blockIdx.x * 256 + threadIdx.x;
  if (i < 1024 * 128) w3p[i] = (_Float16)w3[i];
  if (i < 128 * 64)   w2p[i] = (_Float16)w2[i];
  if (i < 64 * 64)    w1p[i] = (_Float16)w1[i];
  if (i < 64 * 32) {
    int o = i >> 5, k = i & 31;
    w0p[i] = (_Float16)(k < 3 ? w0[o * 3 + k] : 0.0f);
  }
}

// ---------------------------------------------------------------------------
// Kernel 2: sequential sampling (argmax of dist to last chosen point).
// One block per batch. cent layout: [B][M][4] f32 (xyz + pad).
// ---------------------------------------------------------------------------
__global__ void __launch_bounds__(256) fps_kernel(const float* __restrict__ pts,
                                                  float* __restrict__ cent) {
  int b = blockIdx.x;
  const float* P = pts + (size_t)b * NPTS * 3;
  __shared__ float sD[256];
  __shared__ int   sI[256];
  __shared__ float last[3];
  int tid = threadIdx.x;
  if (tid < 3) last[tid] = P[tid];
  if (tid == 0) {
    float* c = cent + (size_t)b * NM * 4;
    c[0] = P[0]; c[1] = P[1]; c[2] = P[2]; c[3] = 0.0f;
  }
  __syncthreads();
  for (int it = 1; it < NM; ++it) {
    float lx = last[0], ly = last[1], lz = last[2];
    float best = -1.0f; int bi = 0;
    for (int p = tid; p < NPTS; p += 256) {
      float dx = P[p * 3 + 0] - lx;
      float dy = P[p * 3 + 1] - ly;
      float dz = P[p * 3 + 2] - lz;
      float d = dx * dx + dy * dy + dz * dz;
      if (d > best) { best = d; bi = p; }
    }
    sD[tid] = best; sI[tid] = bi;
    __syncthreads();
    for (int s = 128; s > 0; s >>= 1) {
      if (tid < s) {
        float d2 = sD[tid + s]; int i2 = sI[tid + s];
        if (d2 > sD[tid] || (d2 == sD[tid] && i2 < sI[tid])) { sD[tid] = d2; sI[tid] = i2; }
      }
      __syncthreads();
    }
    if (tid == 0) {
      int ind = sI[0];
      float x = P[ind * 3 + 0], y = P[ind * 3 + 1], z = P[ind * 3 + 2];
      last[0] = x; last[1] = y; last[2] = z;
      float* c = cent + ((size_t)b * NM + it) * 4;
      c[0] = x; c[1] = y; c[2] = z; c[3] = 0.0f;
    }
    __syncthreads();
  }
}

// ---------------------------------------------------------------------------
// Kernel 3: top-R=32 nearest neighbors per centroid, gathered straight into an
// f16 activation buffer groups[g][r][32] (cols 0..2 = xyz, 3..31 = 0).
// Iterative selection with 64-bit (dist_bits<<32 | idx+1) keys -> stable ties.
// ---------------------------------------------------------------------------
__global__ void __launch_bounds__(256) group_kernel(const float* __restrict__ pts,
                                                    const float* __restrict__ cent,
                                                    _Float16* __restrict__ groups) {
  int g = blockIdx.x;
  int b = g >> 9;
  const float* P = pts + (size_t)b * NPTS * 3;
  const float* c = cent + (size_t)g * 4;
  float cx = c[0], cy = c[1], cz = c[2];
  int tid = threadIdx.x;
  _Float16* G = groups + (size_t)g * NR * 32;
  for (int j = tid; j < NR * 32; j += 256) G[j] = (_Float16)0.0f;
  __shared__ unsigned long long sK[256];
  __shared__ unsigned long long prevKey;
  if (tid == 0) prevKey = 0ull;
  __syncthreads();
  for (int j = 0; j < NR; ++j) {
    unsigned long long pk = prevKey;
    unsigned long long best = ~0ull;
    for (int p = tid; p < NPTS; p += 256) {
      float dx = P[p * 3 + 0] - cx;
      float dy = P[p * 3 + 1] - cy;
      float dz = P[p * 3 + 2] - cz;
      float d = dx * dx + dy * dy + dz * dz;
      unsigned long long key =
          ((unsigned long long)__float_as_uint(d) << 32) | (unsigned)(p + 1);
      if (key > pk && key < best) best = key;
    }
    sK[tid] = best;
    __syncthreads();
    for (int s = 128; s > 0; s >>= 1) {
      if (tid < s) {
        unsigned long long o = sK[tid + s];
        if (o < sK[tid]) sK[tid] = o;
      }
      __syncthreads();
    }
    if (tid == 0) {
      unsigned long long k = sK[0];
      int p = (int)(k & 0xffffffffu) - 1;
      prevKey = k;
      G[j * 32 + 0] = (_Float16)P[p * 3 + 0];
      G[j * 32 + 1] = (_Float16)P[p * 3 + 1];
      G[j * 32 + 2] = (_Float16)P[p * 3 + 2];
    }
    __syncthreads();
  }
}

// ---------------------------------------------------------------------------
// WMMA helpers (CDNA5 v_wmma_f32_16x16x32_f16, wave32)
// ---------------------------------------------------------------------------
__device__ __forceinline__ v8f wmma_f16(v16h a, v16h b, v8f c) {
  return __builtin_amdgcn_wmma_f32_16x16x32_f16(false, a, false, b, (short)0, c,
                                                false, false);
}

// A fragment (16x32 f16), source row-major with row stride ldk (halfs).
__device__ __forceinline__ v16h load_a_frag(const _Float16* act, int lane,
                                            int rowOff, int ldk, int kOff) {
  int r  = (lane & 15) + rowOff;
  int kb = kOff + ((lane >> 4) << 3);
  const _Float16* p = act + r * ldk + kb;
  v8h lo = *(const v8h*)p;
  v8h hi = *(const v8h*)(p + 16);
  v16h out;
#pragma unroll
  for (int i = 0; i < 8; ++i) { out[i] = lo[i]; out[i + 8] = hi[i]; }
  return out;
}

// B fragment (32x16 f16) from weights stored [O][K] row-major (stride ldk).
__device__ __forceinline__ v16h load_b_frag(const _Float16* w, int lane, int ldk,
                                            int nBase, int kOff) {
  int n  = nBase + (lane & 15);
  int kb = kOff + ((lane >> 4) << 3);
  const _Float16* p = w + n * ldk + kb;
  v8h lo = *(const v8h*)p;
  v8h hi = *(const v8h*)(p + 16);
  v16h out;
#pragma unroll
  for (int i = 0; i < 8; ++i) { out[i] = lo[i]; out[i + 8] = hi[i]; }
  return out;
}

// One MLP layer for a 32-row group (A-frags hoisted BEFORE first store, so the
// output region may alias the input region).
template <int KDIM, int ODIM>
__device__ __forceinline__ void mlp_layer(const _Float16* in, int ldkIn,
                                          const _Float16* w,
                                          const float* __restrict__ bias,
                                          const float* __restrict__ sc,
                                          const float* __restrict__ sh,
                                          _Float16* out, int lane) {
  constexpr int NC = KDIM / 32;
  v16h a[2][NC];
#pragma unroll
  for (int t = 0; t < 2; ++t)
#pragma unroll
    for (int c = 0; c < NC; ++c) a[t][c] = load_a_frag(in, lane, t * 16, ldkIn, c * 32);

#pragma unroll
  for (int nt = 0; nt < ODIM / 16; ++nt) {
    v8f acc0 = {}; v8f acc1 = {};
#pragma unroll
    for (int c = 0; c < NC; ++c) {
      v16h bf = load_b_frag(w, lane, KDIM, nt * 16, c * 32);
      acc0 = wmma_f16(a[0][c], bf, acc0);
      acc1 = wmma_f16(a[1][c], bf, acc1);
    }
    int col = nt * 16 + (lane & 15);
    float bb = bias[col], ss = sc[col], tt = sh[col];
    int rb = (lane >> 4) << 3;   // 0 or 8
#pragma unroll
    for (int v = 0; v < 8; ++v) {
      float z0 = fmaxf(0.0f, (acc0[v] + bb) * ss + tt);
      float z1 = fmaxf(0.0f, (acc1[v] + bb) * ss + tt);
      out[(rb + v) * ODIM + col]      = (_Float16)z0;   // rows 0..15
      out[(rb + v + 16) * ODIM + col] = (_Float16)z1;   // rows 16..31
    }
  }
}

// ---------------------------------------------------------------------------
// Async staging of one w3 column tile (16 outs x 128 K = 4096 B, contiguous)
// into LDS: 128 threads x 32 B, via GLOBAL_LOAD_ASYNC_TO_LDS_B128 (ASYNCcnt).
// Builtin signature (from probe diagnostic): param0 = v4i in AS(1) (global),
// param1 = LDS destination, then imm offset + imm cpol.
// ---------------------------------------------------------------------------
typedef __attribute__((address_space(1))) v4i GAS_v4i;
typedef __attribute__((address_space(3))) v4i LAS_v4i;

__device__ __forceinline__ void stage_w3_tile(_Float16* dst,
                                              const _Float16* __restrict__ w3p,
                                              int nt, int tid) {
  const char* src = (const char*)(w3p + (size_t)nt * 2048) + tid * 32;
  char* d = (char*)dst + tid * 32;
#if defined(HAVE_ASYNC_LDS_BUILTIN)
  __builtin_amdgcn_global_load_async_to_lds_b128(
      (GAS_v4i*)src, (LAS_v4i*)d, 0, 0);
  __builtin_amdgcn_global_load_async_to_lds_b128(
      (GAS_v4i*)(src + 16), (LAS_v4i*)(d + 16), 0, 0);
#else
  // inline-asm fallback: VDST = 32-bit LDS byte address, VADDR = 64-bit global
  unsigned l0 = (unsigned)(uintptr_t)d;
  unsigned l1 = l0 + 16;
  unsigned long long g0 = (unsigned long long)(uintptr_t)src;
  unsigned long long g1 = g0 + 16;
  asm volatile("global_load_async_to_lds_b128 %0, %1, off"
               :: "v"(l0), "v"(g0) : "memory");
  asm volatile("global_load_async_to_lds_b128 %0, %1, off"
               :: "v"(l1), "v"(g1) : "memory");
#endif
}

__device__ __forceinline__ void async_wait0() {
#if defined(HAVE_WAIT_ASYNC_BUILTIN)
  __builtin_amdgcn_s_wait_asynccnt(0);
#else
  asm volatile("s_wait_asynccnt 0" ::: "memory");
#endif
}

// ---------------------------------------------------------------------------
// Kernel 4: fused MLP + max-pool. 4 waves/block, one group (32 rows) per wave.
// Last layer: block-shared, double-buffered async staging of w3 tiles.
// ---------------------------------------------------------------------------
__global__ void __launch_bounds__(128) mlp_kernel(
    const _Float16* __restrict__ groups,
    const _Float16* __restrict__ w0p, const _Float16* __restrict__ w1p,
    const _Float16* __restrict__ w2p, const _Float16* __restrict__ w3p,
    const float* __restrict__ b0, const float* __restrict__ s0, const float* __restrict__ t0,
    const float* __restrict__ b1, const float* __restrict__ s1, const float* __restrict__ t1,
    const float* __restrict__ b2, const float* __restrict__ s2, const float* __restrict__ t2,
    const float* __restrict__ b3, const float* __restrict__ s3, const float* __restrict__ t3,
    float* __restrict__ out) {
  // per-wave act area (12 KB): [A: 32x64][B: 32x64]; layer-2 out (32x128) = A+B
  __shared__ __align__(16) _Float16 smem[4 * 6144];     // 48 KB
  __shared__ __align__(16) _Float16 wstage[2][2048];    // 8 KB (w3 tile x2)
  int wave = threadIdx.x >> 5;
  int lane = threadIdx.x & 31;
  int tid  = threadIdx.x;
  int g = blockIdx.x * 4 + wave;

  _Float16* base = smem + wave * 6144;
  _Float16* l0 = base;          // L0 out / L1 in  (32x64)
  _Float16* l1 = base + 2048;   // L1 out / L2 in  (32x64)
  _Float16* l2 = base;          // L2 out / L3 in  (32x128), aliases l0|l1 (safe)
  const _Float16* in0 = groups + (size_t)g * (NR * 32);

  mlp_layer<32, 64>(in0, 32, w0p, b0, s0, t0, l0, lane);
  mlp_layer<64, 64>(l0, 64, w1p, b1, s1, t1, l1, lane);
  mlp_layer<64, 128>(l1, 64, w2p, b2, s2, t2, l2, lane);

  // hoist last-layer A fragments (rows 0..31, K=128)
  v16h a[2][4];
#pragma unroll
  for (int t = 0; t < 2; ++t)
#pragma unroll
    for (int c = 0; c < 4; ++c) a[t][c] = load_a_frag(l2, lane, t * 16, 128, c * 32);

  float* og = out + (size_t)g * 1024;

  stage_w3_tile(wstage[0], w3p, 0, tid);
  async_wait0();
  __syncthreads();

  for (int nt = 0; nt < 64; ++nt) {
    const _Float16* wt = wstage[nt & 1];
    if (nt < 63) stage_w3_tile(wstage[(nt & 1) ^ 1], w3p, nt + 1, tid);

    v8f acc0 = {}; v8f acc1 = {};
#pragma unroll
    for (int c = 0; c < 4; ++c) {
      v16h bf = load_b_frag(wt, lane, 128, 0, c * 32);
      acc0 = wmma_f16(a[0][c], bf, acc0);
      acc1 = wmma_f16(a[1][c], bf, acc1);
    }
    int col = nt * 16 + (lane & 15);
    float bb = b3[col], ss = s3[col], tt = t3[col];
    float mx = -3.0e38f;
#pragma unroll
    for (int v = 0; v < 8; ++v) {
      mx = fmaxf(mx, fmaxf(0.0f, (acc0[v] + bb) * ss + tt));
      mx = fmaxf(mx, fmaxf(0.0f, (acc1[v] + bb) * ss + tt));
    }
    mx = fmaxf(mx, __shfl_xor(mx, 16, 32));   // fold rows held by lane^16
    if (lane < 16) og[col] = mx;

    async_wait0();     // next tile resident
    __syncthreads();   // all waves done reading current tile
  }
}

// ---------------------------------------------------------------------------
// Host launcher
// ---------------------------------------------------------------------------
extern "C" void kernel_launch(void* const* d_in, const int* in_sizes, int n_in,
                              void* d_out, int out_size, void* d_ws, size_t ws_size,
                              hipStream_t stream) {
  const float* points = (const float*)d_in[0];
  const float* w0 = (const float*)d_in[1];
  const float* b0 = (const float*)d_in[2];
  const float* s0 = (const float*)d_in[3];
  const float* t0 = (const float*)d_in[4];
  const float* w1 = (const float*)d_in[5];
  const float* b1 = (const float*)d_in[6];
  const float* s1 = (const float*)d_in[7];
  const float* t1 = (const float*)d_in[8];
  const float* w2 = (const float*)d_in[9];
  const float* b2 = (const float*)d_in[10];
  const float* s2 = (const float*)d_in[11];
  const float* t2 = (const float*)d_in[12];
  const float* w3 = (const float*)d_in[13];
  const float* b3 = (const float*)d_in[14];
  const float* s3 = (const float*)d_in[15];
  const float* t3 = (const float*)d_in[16];
  (void)in_sizes; (void)n_in; (void)out_size; (void)ws_size;

  char* ws = (char*)d_ws;
  float*    cent   = (float*)(ws + 0);                 // [8][512][4] f32   = 65536 B
  _Float16* groups = (_Float16*)(ws + 65536);          // [4096][32][32] f16 = 8388608 B
  _Float16* w0p    = (_Float16*)(ws + 8454144);        // 4096 B
  _Float16* w1p    = (_Float16*)(ws + 8458240);        // 8192 B
  _Float16* w2p    = (_Float16*)(ws + 8466432);        // 16384 B
  _Float16* w3p    = (_Float16*)(ws + 8482816);        // 262144 B

  prep_weights_kernel<<<512, 256, 0, stream>>>(w0, w1, w2, w3, w0p, w1p, w2p, w3p);
  fps_kernel<<<NB, 256, 0, stream>>>(points, cent);
  group_kernel<<<NGRP, 256, 0, stream>>>(points, cent, groups);
  mlp_kernel<<<NGRP / 4, 128, 0, stream>>>(groups, w0p, w1p, w2p, w3p,
                                           b0, s0, t0, b1, s1, t1,
                                           b2, s2, t2, b3, s3, t3,
                                           (float*)d_out);
}
